// RelativeMultiHeadAttention_87531433492934
// MI455X (gfx1250) — matrix operand order repaired and checked by reading, not compiled, not run on hardware
//
#include <hip/hip_runtime.h>
#include <hip/hip_bf16.h>
#include <cstdint>

// ---- problem constants ----
#define L_SEQ  2048
#define DM     1024
#define NH     16
#define HD     64
#define NEMB   1025
#define MAXREL 512

typedef _Float16 half_t;
typedef __attribute__((ext_vector_type(16))) _Float16 v16h;
typedef __attribute__((ext_vector_type(8)))  _Float16 v8h;
typedef __attribute__((ext_vector_type(8)))  float    v8f;

typedef __attribute__((ext_vector_type(4))) unsigned int u32x4;
typedef __attribute__((ext_vector_type(8))) int          i32x8;
typedef __attribute__((ext_vector_type(4))) int          i32x4;

// ---- CDNA5 data-mover feature detection (device pass only) ----
#if defined(__gfx1250__) && __has_builtin(__builtin_amdgcn_global_load_async_to_lds_b128) && __has_builtin(__builtin_amdgcn_s_wait_asynccnt)
#define HAVE_ASYNC 1
#else
#define HAVE_ASYNC 0
#endif

#if defined(__gfx1250__) && __has_builtin(__builtin_amdgcn_tensor_load_to_lds) && __has_builtin(__builtin_amdgcn_s_wait_tensorcnt)
#define HAVE_TDM 1
#else
#define HAVE_TDM 0
#endif

// D = A(16x32 f16) * B(32x16 f16) + C(16x16 f32)
__device__ __forceinline__ v8f wmma32(v16h a, v16h b, v8f c) {
  return __builtin_amdgcn_wmma_f32_16x16x32_f16(false, a, false, b, (short)0, c,
                                                false, false);
}

// A-matrix fragment (16x32, MxK) from row-major f16 buffer (stride in halves).
// Lane L holds row L&15; K chunks [k0+g*8, +8) and [k0+16+g*8, +8), g = L>>4.
__device__ __forceinline__ v16h ldA(const half_t* base, int stride, int lane, int k0) {
  const int row = lane & 15, g = lane >> 4;
  const half_t* p = base + row * stride + k0 + g * 8;
  v8h lo = *(const v8h*)p;
  v8h hi = *(const v8h*)(p + 16);
  return __builtin_shufflevector(lo, hi, 0,1,2,3,4,5,6,7,8,9,10,11,12,13,14,15);
}

// B-matrix fragment (32x16, KxN) from B^T stored row-major (N rows x K halves).
// Lane L holds column L&15; K chunks [k0+g*8, +8) and [k0+16+g*8, +8), g = L>>4.
__device__ __forceinline__ v16h ldB(const half_t* base, int stride, int lane, int k0) {
  const int n = lane & 15, g = lane >> 4;
  const half_t* p = base + n * stride + k0 + g * 8;
  v8h lo = *(const v8h*)p;
  v8h hi = *(const v8h*)(p + 16);
  return __builtin_shufflevector(lo, hi, 0,1,2,3,4,5,6,7,8,9,10,11,12,13,14,15);
}

// ---------------------------------------------------------------------------
// GEMM: C(4096x1024) = A(4096x1024) @ W(1024x1024) + bias
// OUT_MODE 0: f16, heads-split row-major  [bh][l][d]
// OUT_MODE 1: f16, heads-split transposed [bh][d][l]   (for V)
// OUT_MODE 2: f32, plain row-major (final output)
// Block tile 128x64, 8 waves (4x2), each wave 32x32 (2x2 WMMA tiles).
// ---------------------------------------------------------------------------
template <typename AT, int OUT_MODE>
__global__ __launch_bounds__(256) void gemm4096(const AT* __restrict__ A,
                                                const float* __restrict__ W,
                                                const float* __restrict__ bias,
                                                void* __restrict__ outp) {
  __shared__ __align__(16) half_t As[128][40];
  __shared__ __align__(16) half_t Bs[64][40];
  const int tid = threadIdx.x, lane = tid & 31, wid = tid >> 5;
  const int wm = wid & 3, wn = wid >> 2;
  const int mblk = blockIdx.y * 128, nblk = blockIdx.x * 64;
  v8f acc[2][2] = {};
  const int arow = tid >> 1, aseg = (tid & 1) << 4;  // 128 rows x 32 halves
  const int bk = tid >> 3, bn = (tid & 7) << 3;      // transpose-stage W

  for (int kk = 0; kk < DM; kk += 32) {
    __syncthreads();
    {
      const AT* ap = A + (size_t)(mblk + arow) * DM + kk + aseg;
      #pragma unroll
      for (int e = 0; e < 16; ++e) As[arow][aseg + e] = (half_t)ap[e];
    }
    {
      const float* wp = W + (size_t)(kk + bk) * DM + nblk + bn;
      #pragma unroll
      for (int e = 0; e < 8; ++e) Bs[bn + e][bk] = (half_t)wp[e];
    }
    __syncthreads();
    v16h a0 = ldA(&As[wm * 32][0], 40, lane, 0);
    v16h a1 = ldA(&As[wm * 32 + 16][0], 40, lane, 0);
    v16h b0 = ldB(&Bs[wn * 32][0], 40, lane, 0);
    v16h b1 = ldB(&Bs[wn * 32 + 16][0], 40, lane, 0);
    acc[0][0] = wmma32(a0, b0, acc[0][0]);
    acc[0][1] = wmma32(a0, b1, acc[0][1]);
    acc[1][0] = wmma32(a1, b0, acc[1][0]);
    acc[1][1] = wmma32(a1, b1, acc[1][1]);
  }

  const int g = lane >> 4, nlo = lane & 15;
  #pragma unroll
  for (int mi = 0; mi < 2; ++mi)
    #pragma unroll
    for (int ni = 0; ni < 2; ++ni)
      #pragma unroll
      for (int r = 0; r < 8; ++r) {
        const int row = mblk + wm * 32 + mi * 16 + r + 8 * g;
        const int col = nblk + wn * 32 + ni * 16 + nlo;
        const float v = acc[mi][ni][r] + bias[col];
        if constexpr (OUT_MODE == 2) {
          ((float*)outp)[(size_t)row * DM + col] = v;
        } else {
          half_t* o = (half_t*)outp;
          const int b = row >> 11, l = row & (L_SEQ - 1);
          const int h = col >> 6, d = col & (HD - 1);
          if constexpr (OUT_MODE == 0)
            o[(((size_t)(b * NH + h)) * L_SEQ + l) * HD + d] = (half_t)v;
          else
            o[(((size_t)(b * NH + h)) * HD + d) * L_SEQ + l] = (half_t)v;
        }
      }
}

// ---------------------------------------------------------------------------
// q_rel(65536 x 1025) = Qh(65536 x 64) @ rel_emb^T(64 x 1025), f16 out.
// rel_emb (1025 x 64) row-major IS already the B^T layout we need.
// ---------------------------------------------------------------------------
__global__ __launch_bounds__(256) void qrel_gemm(const half_t* __restrict__ Qh,
                                                 const float* __restrict__ E,
                                                 half_t* __restrict__ qrel) {
  __shared__ __align__(16) half_t Bs[64][72];
  const int tid = threadIdx.x, lane = tid & 31, wid = tid >> 5;
  const int n0 = blockIdx.x * 64;
  const int m0w = blockIdx.y * 128 + wid * 16;
  {
    const int r = tid >> 2, c0 = (tid & 3) << 4;
    const int n = n0 + r;
    const float* ep = E + (size_t)n * HD + c0;
    #pragma unroll
    for (int e = 0; e < 16; ++e)
      Bs[r][c0 + e] = (n < NEMB) ? (half_t)ep[e] : (half_t)0;
  }
  __syncthreads();
  v16h a0 = ldA(Qh + (size_t)m0w * HD, HD, lane, 0);
  v16h a1 = ldA(Qh + (size_t)m0w * HD, HD, lane, 32);
  v8f acc[4] = {};
  #pragma unroll
  for (int ni = 0; ni < 4; ++ni) {
    v16h b0 = ldB(&Bs[ni * 16][0], 72, lane, 0);
    acc[ni] = wmma32(a0, b0, acc[ni]);
    v16h b1 = ldB(&Bs[ni * 16][0], 72, lane, 32);
    acc[ni] = wmma32(a1, b1, acc[ni]);
  }
  const int g = lane >> 4, nlo = lane & 15;
  #pragma unroll
  for (int ni = 0; ni < 4; ++ni)
    #pragma unroll
    for (int r = 0; r < 8; ++r) {
      const int col = n0 + ni * 16 + nlo;
      if (col < NEMB) {
        const int row = m0w + r + 8 * g;
        qrel[(size_t)row * NEMB + col] = (half_t)acc[ni][r];
      }
    }
}

// ---------------------------------------------------------------------------
// Flash-style attention per (b,h): streaming softmax over j-blocks of 64.
// grid = (L/128, B*H); 8 waves x 16 rows. K tile is moved by the Tensor Data
// Mover (1 descriptor / tile; pad feature creates the 144B LDS row stride in
// hardware); V tile streams via GLOBAL_LOAD_ASYNC_TO_LDS_B128 (ASYNCcnt).
// ---------------------------------------------------------------------------
__global__ __launch_bounds__(256) void attn_kernel(const half_t* __restrict__ Qh,
                                                   const half_t* __restrict__ Kh,
                                                   const half_t* __restrict__ Vt,
                                                   const half_t* __restrict__ qrel,
                                                   half_t* __restrict__ Om) {
  __shared__ __align__(16) half_t Ks[64][72];       // K rows (j, d)  == B^T
  __shared__ __align__(16) half_t Vs[64][72];       // V^T rows (d, j) == B^T
  __shared__ __align__(16) half_t Ps[8][16][72];    // per-wave P bounce buffer
  const int tid = threadIdx.x, lane = tid & 31, wid = tid >> 5;
  const int g = lane >> 4, nlo = lane & 15;
  const int bh = blockIdx.y;
  const int i0w = blockIdx.x * 128 + wid * 16;

  const half_t* Qb = Qh + (size_t)bh * L_SEQ * HD;
  const half_t* Kb = Kh + (size_t)bh * L_SEQ * HD;
  const half_t* Vb = Vt + (size_t)bh * HD * L_SEQ;
  const half_t* qr = qrel + (size_t)bh * L_SEQ * NEMB;

  const v16h aq0 = ldA(Qb + (size_t)i0w * HD, HD, lane, 0);
  const v16h aq1 = ldA(Qb + (size_t)i0w * HD, HD, lane, 32);

  float m_r[8], l_r[8];
  v8f o_acc[4] = {};
  #pragma unroll
  for (int r = 0; r < 8; ++r) { m_r[r] = -1e30f; l_r[r] = 0.f; }

  for (int jb = 0; jb < L_SEQ; jb += 64) {
    __syncthreads();
    {
      const int jl = tid >> 2, c0 = (tid & 3) << 4;
#if HAVE_TDM
      if (wid == 0) {
        // --- TDM: K tile (64 rows x 128B, contiguous in global) -> Ks ---
        // D# group0: count=1 | lds_addr | global_addr[56:0] | type=2
        const unsigned long long gk =
            (unsigned long long)(uintptr_t)(Kb + (size_t)jb * HD);
        const unsigned lk = (unsigned)(uintptr_t)&Ks[0][0];
        u32x4 g0 = {1u, lk, (unsigned)gk,
                    (unsigned)(gk >> 32) | (2u << 30)};
        // group1: data_size=2B(code1), pad_enable, pad every 32 DW (+4 DW):
        // LDS row stride 128+16 = 144B = 72 halves. tensor/tile = 64x64.
        i32x8 g1 = {(int)((1u << 16) | (1u << 20) | (4u << 22) | (3u << 25)),
                    (int)(64u << 16),  // tensor_dim0 = 64  (bits 79:48)
                    (int)(64u << 16),  // tensor_dim1 = 64  (bits 111:80)
                    (int)(64u << 16),  // tile_dim0   = 64  (bits 127:112)
                    64,                // tile_dim1   = 64
                    64,                // tensor_dim0_stride = 64 elements
                    0, 0};
        i32x4 z4 = {0, 0, 0, 0};
        i32x8 z8 = {0, 0, 0, 0, 0, 0, 0, 0};
        __builtin_amdgcn_tensor_load_to_lds(g0, g1, z4, z4, z8, 0);
      }
#elif HAVE_ASYNC
      const half_t* kp = Kb + (size_t)(jb + jl) * HD + c0;
      half_t* lk = &Ks[jl][c0];
      __builtin_amdgcn_global_load_async_to_lds_b128((i32x4*)kp, (i32x4*)lk, 0, 0);
      __builtin_amdgcn_global_load_async_to_lds_b128((i32x4*)(kp + 8), (i32x4*)(lk + 8), 0, 0);
#else
      const half_t* kp = Kb + (size_t)(jb + jl) * HD + c0;
      v8h* dk = (v8h*)&Ks[jl][c0];
      dk[0] = ((const v8h*)kp)[0];
      dk[1] = ((const v8h*)kp)[1];
#endif
      const half_t* vp = Vb + (size_t)jl * L_SEQ + jb + c0;  // jl == d here
#if HAVE_ASYNC
      half_t* lv = &Vs[jl][c0];
      __builtin_amdgcn_global_load_async_to_lds_b128((i32x4*)vp, (i32x4*)lv, 0, 0);
      __builtin_amdgcn_global_load_async_to_lds_b128((i32x4*)(vp + 8), (i32x4*)(lv + 8), 0, 0);
#else
      v8h* dv = (v8h*)&Vs[jl][c0];
      dv[0] = ((const v8h*)vp)[0];
      dv[1] = ((const v8h*)vp)[1];
#endif
    }
#if HAVE_TDM
    if (wid == 0) __builtin_amdgcn_s_wait_tensorcnt(0);
#endif
#if HAVE_ASYNC
    __builtin_amdgcn_s_wait_asynccnt(0);
#endif
    __syncthreads();

    // S = Q K^T (16 x 64 per wave)
    v8f s[4] = {};
    #pragma unroll
    for (int ni = 0; ni < 4; ++ni) {
      v16h b0 = ldB(&Ks[ni * 16][0], 72, lane, 0);
      s[ni] = wmma32(aq0, b0, s[ni]);
      v16h b1 = ldB(&Ks[ni * 16][0], 72, lane, 32);
      s[ni] = wmma32(aq1, b1, s[ni]);
    }

    // + relative position gather, * 1/sqrt(d); per-row running max
    float rm[8];
    #pragma unroll
    for (int r = 0; r < 8; ++r) rm[r] = -1e30f;
    #pragma unroll
    for (int ni = 0; ni < 4; ++ni)
      #pragma unroll
      for (int r = 0; r < 8; ++r) {
        const int i = i0w + r + 8 * g;
        const int j = jb + ni * 16 + nlo;
        int rel = j - i;
        rel = rel < -MAXREL ? -MAXREL : (rel > MAXREL ? MAXREL : rel);
        const float v =
            (s[ni][r] + (float)qr[(size_t)i * NEMB + rel + MAXREL]) * 0.125f;
        s[ni][r] = v;
        rm[r] = fmaxf(rm[r], v);
      }
    #pragma unroll
    for (int r = 0; r < 8; ++r) {
      float v = rm[r];
      v = fmaxf(v, __shfl_xor(v, 1, 16));
      v = fmaxf(v, __shfl_xor(v, 2, 16));
      v = fmaxf(v, __shfl_xor(v, 4, 16));
      v = fmaxf(v, __shfl_xor(v, 8, 16));
      rm[r] = v;
    }

    // online softmax update
    float alpha[8], rs[8];
    #pragma unroll
    for (int r = 0; r < 8; ++r) {
      const float mn = fmaxf(m_r[r], rm[r]);
      alpha[r] = __expf(m_r[r] - mn);
      m_r[r] = mn;
      rs[r] = 0.f;
    }
    #pragma unroll
    for (int ni = 0; ni < 4; ++ni)
      #pragma unroll
      for (int r = 0; r < 8; ++r) {
        const float p = __expf(s[ni][r] - m_r[r]);
        s[ni][r] = p;
        rs[r] += p;
      }
    #pragma unroll
    for (int r = 0; r < 8; ++r) {
      float v = rs[r];
      v += __shfl_xor(v, 1, 16);
      v += __shfl_xor(v, 2, 16);
      v += __shfl_xor(v, 4, 16);
      v += __shfl_xor(v, 8, 16);
      l_r[r] = l_r[r] * alpha[r] + v;
    }
    #pragma unroll
    for (int di = 0; di < 4; ++di)
      #pragma unroll
      for (int r = 0; r < 8; ++r) o_acc[di][r] *= alpha[r];

    // C-layout -> A-layout bounce of P through per-wave LDS slab
    #pragma unroll
    for (int ni = 0; ni < 4; ++ni)
      #pragma unroll
      for (int r = 0; r < 8; ++r)
        Ps[wid][r + 8 * g][ni * 16 + nlo] = (half_t)s[ni][r];
    // same-wave LDS ops are in-order (DScnt), no barrier needed
    v16h ap0 = ldA(&Ps[wid][0][0], 72, lane, 0);
    v16h ap1 = ldA(&Ps[wid][0][0], 72, lane, 32);

    // O += P V   (B^T layout = V^T rows = Vs)
    #pragma unroll
    for (int di = 0; di < 4; ++di) {
      v16h bv0 = ldB(&Vs[di * 16][0], 72, lane, 0);
      o_acc[di] = wmma32(ap0, bv0, o_acc[di]);
      v16h bv1 = ldB(&Vs[di * 16][0], 72, lane, 32);
      o_acc[di] = wmma32(ap1, bv1, o_acc[di]);
    }
  }

  // epilogue: normalize and store merged (b, l, h*64+d) f16
  const int b = bh >> 4, h = bh & 15;
  float inv[8];
  #pragma unroll
  for (int r = 0; r < 8; ++r) inv[r] = 1.0f / l_r[r];
  #pragma unroll
  for (int di = 0; di < 4; ++di)
    #pragma unroll
    for (int r = 0; r < 8; ++r) {
      const int i = i0w + r + 8 * g;
      const int d = di * 16 + nlo;
      Om[((size_t)(b * L_SEQ) + i) * DM + h * HD + d] =
          (half_t)(o_acc[di][r] * inv[r]);
    }
}

// ---------------------------------------------------------------------------
extern "C" void kernel_launch(void* const* d_in, const int* in_sizes, int n_in,
                              void* d_out, int out_size, void* d_ws, size_t ws_size,
                              hipStream_t stream) {
  (void)in_sizes; (void)n_in; (void)out_size; (void)ws_size;
  const float* query = (const float*)d_in[0];
  const float* key   = (const float*)d_in[1];
  const float* value = (const float*)d_in[2];
  const float* wq = (const float*)d_in[3];
  const float* bq = (const float*)d_in[4];
  const float* wk = (const float*)d_in[5];
  const float* bk = (const float*)d_in[6];
  const float* wv = (const float*)d_in[7];
  const float* bv = (const float*)d_in[8];
  const float* wo = (const float*)d_in[9];
  const float* bo = (const float*)d_in[10];
  const float* rel = (const float*)d_in[11];

  // workspace (f16): Qh, Kh, Vt, Om: 4 x 4,194,304 halves; qrel: 67,174,400 halves
  half_t* ws = (half_t*)d_ws;
  const size_t QKV = (size_t)2 * NH * L_SEQ * HD;  // 4,194,304
  half_t* Qh = ws;
  half_t* Kh = Qh + QKV;
  half_t* Vt = Kh + QKV;
  half_t* Om = Vt + QKV;
  half_t* qrel = Om + QKV;

  dim3 blk(256), g1(DM / 64, (2 * L_SEQ) / 128);  // (16, 32)
  gemm4096<float, 0><<<g1, blk, 0, stream>>>(query, wq, bq, Qh);
  gemm4096<float, 0><<<g1, blk, 0, stream>>>(key, wk, bk, Kh);
  gemm4096<float, 1><<<g1, blk, 0, stream>>>(value, wv, bv, Vt);
  qrel_gemm<<<dim3(17, 512), blk, 0, stream>>>(Qh, rel, qrel);
  attn_kernel<<<dim3(L_SEQ / 128, 2 * NH), blk, 0, stream>>>(Qh, Kh, Vt, qrel, Om);
  gemm4096<half_t, 2><<<g1, blk, 0, stream>>>(Om, wo, bo, d_out);
}
